// TestSeqNMSModule_32779190403243
// MI455X (gfx1250) — 8-bit matrix operand order repaired and checked by reading, compiled, not run on hardware
//
#include <hip/hip_runtime.h>
#include <hip/hip_bf16.h>


#define F_FRAMES 128
#define N_BOX    256
#define LINK_THR 0.2f
#define IOU_THR  0.2f
#define MAX_ITERS 512

typedef int v8i __attribute__((ext_vector_type(8)));

__global__ __launch_bounds__(256, 4)
void build_adj_kernel(const float* __restrict__ boxes,
                      const int* __restrict__ classes,
                      unsigned* __restrict__ adj)
{
  const int lane = threadIdx.x & 31;
  const int wid  = blockIdx.x * (blockDim.x >> 5) + (threadIdx.x >> 5);
  const int f    = wid >> 7;
  const int rest = wid & 127;
  const int it   = rest >> 3;
  const int jb   = rest & 7;
  const int i0   = it << 4;
  const int half = (lane >> 4) & 1;
  const int l15  = lane & 15;

  const int cA = classes[f * N_BOX + i0 + l15];

  unsigned bal[2][8];

#pragma unroll
  for (int jt = 0; jt < 2; ++jt) {
    const int j0 = (jb << 5) + (jt << 4);
    const int cB = classes[(f + 1) * N_BOX + j0 + l15];

    v8i acc = {};
#pragma unroll
    for (int kc = 0; kc < 2; ++kc) {
      const int kofs = kc << 6;
      v8i afrag, bfrag;
#pragma unroll
      for (int v = 0; v < 8; ++v) {
        const int ka = kofs + half * 16 + (v & 3) * 4 + ((v >> 2) & 1) * 32;
        unsigned aw = 0;
#pragma unroll
        for (int b = 0; b < 4; ++b)
          aw |= (cA == (ka + b)) ? (1u << (8 * b)) : 0u;
        afrag[v] = (int)aw;
        const int kb = kofs + half * 16 + (v & 3) * 4 + ((v >> 2) & 1) * 32;
        unsigned bw = 0;
#pragma unroll
        for (int b = 0; b < 4; ++b)
          bw |= (cB == (kb + b)) ? (1u << (8 * b)) : 0u;
        bfrag[v] = (int)bw;
      }
      acc = __builtin_amdgcn_wmma_i32_16x16x64_iu8(false, afrag, false, bfrag, acc,
                                                   false, false);
    }

    const float4 cbx = ((const float4*)boxes)[(f + 1) * N_BOX + j0 + l15];
    const float areaB = (cbx.z - cbx.x) * (cbx.w - cbx.y);
#pragma unroll
    for (int v = 0; v < 8; ++v) {
      const int row = v + half * 8;
      const float4 rb = ((const float4*)boxes)[f * N_BOX + i0 + row];
      float lx = fmaxf(rb.x, cbx.x), ly = fmaxf(rb.y, cbx.y);
      float rx = fminf(rb.z, cbx.z), ry = fminf(rb.w, cbx.w);
      float iw = fmaxf(rx - lx, 0.f), ih = fmaxf(ry - ly, 0.f);
      float inter = iw * ih;
      float areaA = (rb.z - rb.x) * (rb.w - rb.y);
      float iou = inter / fmaxf(areaA + areaB - inter, 1e-12f);
      bool pred = (acc[v] > 0) && (iou >= LINK_THR);
      bal[jt][v] = (unsigned)__ballot((int)pred);
    }
  }

  if (lane == 0) {
#pragma unroll
    for (int v = 0; v < 8; ++v) {
      unsigned lo0 = bal[0][v] & 0xffffu, hi0 = (bal[0][v] >> 16) & 0xffffu;
      unsigned lo1 = bal[1][v] & 0xffffu, hi1 = (bal[1][v] >> 16) & 0xffffu;
      adj[(f * N_BOX + i0 + v)     * 8 + jb] = lo0 | (lo1 << 16);
      adj[(f * N_BOX + i0 + v + 8) * 8 + jb] = hi0 | (hi1 << 16);
    }
  }
}

#define SM_SC    0
#define SM_NXT   131072
#define SM_REM   196608
#define SM_ABUF  200704
#define SM_MS    217088
#define SM_LN    219136
#define SM_REDV  221184
#define SM_REDK  225280
#define SM_REDL  229376
#define SM_SEQ   233472
#define SM_TOTAL 234496

__global__ __launch_bounds__(1024, 1)
void seqnms_kernel(const float* __restrict__ boxes,
                   const float* __restrict__ scores,
                   const unsigned* __restrict__ adj,
                   float* __restrict__ out)
{
  extern __shared__ __align__(16) char smem[];
  float*    sc   = (float*)(smem + SM_SC);
  short*    nxt  = (short*)(smem + SM_NXT);
  unsigned* rem  = (unsigned*)(smem + SM_REM);
  unsigned* abuf = (unsigned*)(smem + SM_ABUF);
  float*    msb  = (float*)(smem + SM_MS);
  float*    lnb  = (float*)(smem + SM_LN);
  float*    redv = (float*)(smem + SM_REDV);
  int*      redk = (int*)(smem + SM_REDK);
  float*    redl = (float*)(smem + SM_REDL);
  int*      seq  = (int*)(smem + SM_SEQ);

  __shared__ float s_avg, s_L;
  __shared__ int   s_kstar;

  const int t    = threadIdx.x;
  const int i    = t >> 2;
  const int part = t & 3;

  for (int e = t; e < F_FRAMES * N_BOX; e += 1024) sc[e] = scores[e];
  rem[t] = 0u;
  if (t < N_BOX) nxt[127 * N_BOX + t] = -1;
  __syncthreads();

  for (int iter = 0; iter < MAX_ITERS; ++iter) {
    float bestv = -1.0f; int bestk = 0x7fffffff; float bestl = 1.0f;
    if (part == 0) {
      float s = sc[127 * N_BOX + i];
      msb[i] = s; lnb[i] = 1.0f;
      bestv = s; bestk = 127 * N_BOX + i; bestl = 1.0f;
    }
    if (t < 512) {
      unsigned lds = (unsigned)(size_t)abuf + (unsigned)(t * 16);
      unsigned long long ga = (unsigned long long)(size_t)(adj + 126 * 2048)
                            + (unsigned long long)(t * 16);
      asm volatile("global_load_async_to_lds_b128 %0, %1, off"
                   :: "v"(lds), "v"(ga) : "memory");
    }

    int cb = 0, pb = 0;
    for (int f = 126; f >= 0; --f) {
      asm volatile("s_wait_asynccnt 0x0" ::: "memory");
      __syncthreads();
      if (f > 0 && t < 512) {
        unsigned lds = (unsigned)(size_t)abuf + (unsigned)((cb ^ 1) * 8192 + t * 16);
        unsigned long long ga = (unsigned long long)(size_t)(adj + (f - 1) * 2048)
                              + (unsigned long long)(t * 16);
        asm volatile("global_load_async_to_lds_b128 %0, %1, off"
                     :: "v"(lds), "v"(ga) : "memory");
      }

      const unsigned* arow = abuf + cb * 2048 + i * 8;
      const unsigned* remN = rem + (f + 1) * 8;
      const unsigned rowSup = (rem[f * 8 + (i >> 5)] >> (i & 31)) & 1u;
      float best = 0.0f; int arg = -1;
#pragma unroll
      for (int wi = 0; wi < 2; ++wi) {
        int w = part * 2 + wi;
        unsigned m = arow[w] & ~remN[w];
        m = rowSup ? 0u : m;
        int base = w << 5;
        while (m) {
          int b = __builtin_ctz(m);
          m &= m - 1;
          float v = msb[pb * N_BOX + base + b];
          if (v > best) { best = v; arg = base + b; }
        }
      }
#pragma unroll
      for (int off = 1; off < 4; off <<= 1) {
        float ov = __shfl_down(best, off, 4);
        int   oa = __shfl_down(arg,  off, 4);
        if (ov > best) { best = ov; arg = oa; }
      }
      if (part == 0) {
        bool  has = best > 0.0f;
        float ms  = sc[f * N_BOX + i] + best;
        float ln  = 1.0f + (has ? lnb[pb * N_BOX + arg] : 0.0f);
        msb[(pb ^ 1) * N_BOX + i] = ms;
        lnb[(pb ^ 1) * N_BOX + i] = ln;
        nxt[f * N_BOX + i] = (short)(has ? arg : -1);
        if (ms >= bestv) { bestv = ms; bestk = f * N_BOX + i; bestl = ln; }
      }
      cb ^= 1; pb ^= 1;
    }

    redv[t] = bestv; redk[t] = bestk; redl[t] = bestl;
    __syncthreads();
    for (int s = 512; s >= 1; s >>= 1) {
      if (t < s) {
        float v2 = redv[t + s]; int k2 = redk[t + s];
        if (v2 > redv[t] || (v2 == redv[t] && k2 < redk[t])) {
          redv[t] = v2; redk[t] = k2; redl[t] = redl[t + s];
        }
      }
      __syncthreads();
    }
    if (t == 0) { s_kstar = redk[0]; s_L = redl[0]; }
    __syncthreads();

    if (s_L <= 1.0f) break;

    if (t < F_FRAMES) seq[t] = -1;
    __syncthreads();
    if (t == 0) {
      int   fcur = s_kstar >> 8;
      int   cur  = s_kstar & 255;
      float sum  = 0.0f;
      while (cur >= 0 && fcur < F_FRAMES) {
        seq[fcur] = cur;
        sum += sc[fcur * N_BOX + cur];
        cur = (int)nxt[fcur * N_BOX + cur];
        ++fcur;
      }
      s_avg = sum / fmaxf(s_L, 1.0f);
    }
    __syncthreads();

    if (t < F_FRAMES && seq[t] >= 0) sc[t * N_BOX + seq[t]] = s_avg;

    for (int e = t; e < F_FRAMES * N_BOX; e += 1024) {
      int fr = e >> 8, j = e & 255;
      int si = seq[fr];
      bool pred = false;
      if (si >= 0) {
        float4 sb = ((const float4*)boxes)[fr * N_BOX + si];
        float4 bb = ((const float4*)boxes)[fr * N_BOX + j];
        float lx = fmaxf(sb.x, bb.x), ly = fmaxf(sb.y, bb.y);
        float rx = fminf(sb.z, bb.z), ry = fminf(sb.w, bb.w);
        float iw = fmaxf(rx - lx, 0.f), ih = fmaxf(ry - ly, 0.f);
        float inter = iw * ih;
        float areaS = (sb.z - sb.x) * (sb.w - sb.y);
        float areaB = (bb.z - bb.x) * (bb.w - bb.y);
        float iou = inter / fmaxf(areaS + areaB - inter, 1e-12f);
        pred = iou >= IOU_THR;
      }
      unsigned word = (unsigned)__ballot((int)pred);
      if ((t & 31) == 0 && word) rem[(fr << 3) + (j >> 5)] |= word;
    }
    __syncthreads();
  }

  for (int e = t; e < F_FRAMES * N_BOX; e += 1024) out[e] = sc[e];
}

extern "C" void kernel_launch(void* const* d_in, const int* in_sizes, int n_in,
                              void* d_out, int out_size, void* d_ws, size_t ws_size,
                              hipStream_t stream) {
  (void)in_sizes; (void)n_in; (void)out_size; (void)ws_size;
  const float* boxes   = (const float*)d_in[0];
  const float* scores  = (const float*)d_in[1];
  const int*   classes = (const int*)d_in[2];
  float*       out     = (float*)d_out;
  unsigned*    adj     = (unsigned*)d_ws;

  build_adj_kernel<<<2032, 256, 0, stream>>>(boxes, classes, adj);
  seqnms_kernel<<<1, 1024, SM_TOTAL, stream>>>(boxes, scores, adj, out);
}
